// ResNetBlock_MoE_8091718385701
// MI455X (gfx1250) — matrix operand order repaired and checked by reading, not compiled, not run on hardware
//
#include <hip/hip_runtime.h>
#include <hip/hip_bf16.h>

// ---------------------------------------------------------------------------
// MoE ResNet BasicBlock for MI455X (gfx1250), wave32 + WMMA bf16 path.
// B=64, C=64, H=W=56, E=8, TOPK=2.
// ---------------------------------------------------------------------------

typedef __attribute__((ext_vector_type(16))) __bf16 v16bf;
typedef __attribute__((ext_vector_type(8)))  __bf16 v8bf;
typedef __attribute__((ext_vector_type(2)))  __bf16 v2bf;
typedef __attribute__((ext_vector_type(8)))  float  v8f;

#define BATCH   64
#define CH      64
#define HH      56
#define WW      56
#define HWSZ    (HH * WW)          // 3136
#define NEXP    8
#define TOPK    2
#define TILES   28                 // 28 tiles of 2 output rows each
#define OUT_ELEMS ((size_t)BATCH * CH * HWSZ)   // 12,845,056

// workspace layout
#define WPACK_ELEMS (NEXP * 2 * 4 * 9 * 2 * 512)          // 589,824 bf16
#define WPACK_BYTES (WPACK_ELEMS * 2)                     // 1,179,648 B
#define Y1_ELEMS    ((size_t)BATCH * TOPK * HWSZ * CH)    // 25,690,112 bf16
#define Y1_BYTES    (Y1_ELEMS * 2)                        // 51,380,224 B

// LDS input tile: 4 halo rows x 58 halo cols, channel stride 80 (pad for banks,
// keeps 32B alignment for v16bf loads)
#define LDS_CSTR 80
#define LDS_ELEMS (4 * 58 * LDS_CSTR)

// ---------------------------------------------------------------------------
// Gate: GAP -> linear(8) -> top-2 -> softmax.  dense_w to d_out tail.
// ---------------------------------------------------------------------------
__global__ __launch_bounds__(256) void gate_kernel(
    const float* __restrict__ x, const float* __restrict__ gw,
    const float* __restrict__ gb, float* __restrict__ dense_out,
    int* __restrict__ sel_i, float* __restrict__ sel_w)
{
  __shared__ float red[256];
  __shared__ float pooled[CH];
  __shared__ float lg[NEXP];

  const int b   = blockIdx.x;
  const int tid = threadIdx.x;
  const int c   = tid & 63;
  const int grp = tid >> 6;      // 0..3

  const float* xb = x + ((size_t)b * CH + c) * HWSZ;
  float s = 0.f;
  for (int p = grp; p < HWSZ; p += 4) s += xb[p];
  red[tid] = s;
  __syncthreads();
  if (tid < CH) {
    pooled[tid] = (red[tid] + red[tid + 64] + red[tid + 128] + red[tid + 192])
                  * (1.0f / (float)HWSZ);
  }
  __syncthreads();
  if (tid < NEXP) {
    float acc = gb[tid];
    for (int cc = 0; cc < CH; ++cc) acc += pooled[cc] * gw[tid * CH + cc];
    lg[tid] = acc;
  }
  __syncthreads();
  if (tid == 0) {
    int i0 = 0;
    for (int e = 1; e < NEXP; ++e) if (lg[e] > lg[i0]) i0 = e;
    int i1 = (i0 == 0) ? 1 : 0;
    for (int e = 0; e < NEXP; ++e)
      if (e != i0 && lg[e] > lg[i1]) i1 = e;
    float v0 = lg[i0], v1 = lg[i1];
    float e1 = __expf(v1 - v0);          // v0 is the max
    float w0 = 1.0f / (1.0f + e1);
    float w1 = e1 / (1.0f + e1);
    for (int e = 0; e < NEXP; ++e) {
      float v = (e == i0) ? w0 : ((e == i1) ? w1 : 0.0f);
      dense_out[b * NEXP + e] = v;
    }
    sel_i[b * 2 + 0] = i0;  sel_i[b * 2 + 1] = i1;
    sel_w[b * 2 + 0] = w0;  sel_w[b * 2 + 1] = w1;
  }
}

// ---------------------------------------------------------------------------
// Repack OIHW fp32 conv weights into bf16 WMMA A-fragments.
// Fragment f = (((e*2 + layer)*4 + m)*9 + tap)*2 + q ; 512 bf16 = 32 lanes x 16.
// A 16x32 layout: lane L<16 -> M=L, K = 0..15 ; lane L>=16 -> M=L-16,
// K = 16..31.  K = q*32 + k, where k indexes C_in within the chunk.
// ---------------------------------------------------------------------------
__global__ __launch_bounds__(256) void repack_kernel(
    const float* __restrict__ w1, const float* __restrict__ w2,
    __bf16* __restrict__ wp)
{
  int idx = blockIdx.x * 256 + threadIdx.x;
  if (idx >= WPACK_ELEMS) return;
  int i = idx & 15;
  int L = (idx >> 4) & 31;
  int f = idx >> 9;
  int q = f & 1;
  int t = (f >> 1) % 9;
  int m = (f / 18) & 3;
  int l = (f / 72) & 1;
  int e = f / 144;
  int r = t / 3, s = t % 3;
  int co = m * 16 + (L & 15);
  int kb = (L < 16) ? 0 : 16;
  int k  = kb + i;
  int ci = q * 32 + k;
  const float* w = l ? w2 : w1;
  float v = w[((((size_t)e * CH + co) * CH + ci) * 3 + r) * 3 + s];
  wp[idx] = (__bf16)v;
}

// ---------------------------------------------------------------------------
// Implicit-GEMM 3x3 conv, M=C_out=64, N=112 positions (2 rows, padded to 128),
// K = 9 taps x 64 C_in.  8 waves: 4 co-blocks x 2 spatial halves; each wave
// owns 4 WMMA N-tiles.  Fully unrolled tap loop -> constant DS offsets.
// LAYER=1: x(fp32 NCHW) -> relu(bn1(conv1)) -> y1 (bf16 NHWC).
// LAYER=2: y1 -> bn2(conv2) + x, relu, gate-weighted sum over k -> out (NCHW).
// ---------------------------------------------------------------------------
template<int LAYER>
__global__ __launch_bounds__(256) void moe_conv_kernel(
    const float* __restrict__ x, const float* __restrict__ bn_s,
    const float* __restrict__ bn_b, const __bf16* __restrict__ wpack,
    __bf16* __restrict__ y1, const int* __restrict__ sel_i,
    const float* __restrict__ sel_wt, float* __restrict__ out)
{
  __shared__ __attribute__((aligned(32))) __bf16 lds[LDS_ELEMS];

  const int tid  = threadIdx.x;
  const int lane = tid & 31;
  const int wave = tid >> 5;
  const int m    = wave & 3;             // co block (16 rows)
  const int g    = wave >> 2;            // spatial half (4 n-tiles each)
  const int nl   = lane & 15;            // N within a WMMA tile
  const int chHalf = (lane >> 4) << 4;   // B-frag K-half: 0 or 16
  const int cob  = m * 16 + ((lane >> 4) << 3);  // first C_out of this lane's C/D

  int tile, b, k0, nK;
  if (LAYER == 1) {
    tile = blockIdx.x % TILES;
    int bk = blockIdx.x / TILES;
    b = bk >> 1; k0 = bk & 1; nK = 1;
  } else {
    tile = blockIdx.x % TILES;
    b = blockIdx.x / TILES;
    k0 = 0; nK = 2;
  }
  const int h0 = tile * 2;

  // ---- per-lane position decode, shared by all phases ----
  int  pRow[4], pCol[4], ldsBase[4];
  bool pVal[4];
#pragma unroll
  for (int jj = 0; jj < 4; ++jj) {
    int p = (g * 4 + jj) * 16 + nl;
    pVal[jj] = p < 112;
    int pp = pVal[jj] ? p : 111;          // padded lanes compute garbage
    pRow[jj] = pp / WW;
    pCol[jj] = pp % WW;
    ldsBase[jj] = (pRow[jj] * 58 + pCol[jj]) * LDS_CSTR + chHalf;
  }

  float xres[4][8];
  float res[4][8];
  if (LAYER == 2) {
#pragma unroll
    for (int jj = 0; jj < 4; ++jj) {
      int hw = (h0 + pRow[jj]) * WW + pCol[jj];
#pragma unroll
      for (int vg = 0; vg < 8; ++vg) {
        xres[jj][vg] = pVal[jj] ? x[((size_t)b * CH + cob + vg) * HWSZ + hw] : 0.f;
        res[jj][vg] = 0.f;
      }
    }
  }

  for (int k = k0; k < k0 + nK; ++k) {
    const int e = sel_i[b * 2 + k];

    // ---- stage halo tile (4 rows x 58 cols x 64 ch) into LDS, pos-major ----
    __syncthreads();
    if (LAYER == 1) {
      const float* xb = x + (size_t)b * CH * HWSZ;
      // 2 channels packed per 4B DS store; shift/mask decode only.
      for (int idx = tid; idx < 4 * 64 * 32; idx += 256) {
        int col = idx & 63;                 // padded to 64; skip >= 58
        int cp  = (idx >> 6) & 31;          // channel pair
        int row = idx >> 11;                // 0..3
        if (col < 58) {
          int h = h0 - 1 + row;
          int w = col - 1;
          float v0 = 0.f, v1 = 0.f;
          if ((unsigned)h < (unsigned)HH && (unsigned)w < (unsigned)WW) {
            const float* p0 = xb + (size_t)(cp * 2) * HWSZ + h * WW + w;
            v0 = p0[0];
            v1 = p0[HWSZ];
          }
          v2bf pk;
          pk[0] = (__bf16)v0;
          pk[1] = (__bf16)v1;
          *(v2bf*)&lds[(row * 58 + col) * LDS_CSTR + cp * 2] = pk;
        }
      }
    } else {
      const __bf16* yb = y1 + (size_t)(b * TOPK + k) * HWSZ * CH;
      // straight 16B copies: NHWC bf16 is channel-contiguous.
      for (int idx = tid; idx < 4 * 64 * 8; idx += 256) {
        int co8 = idx & 7;                  // channel octet
        int col = (idx >> 3) & 63;          // padded to 64; skip >= 58
        int row = idx >> 9;                 // 0..3
        if (col < 58) {
          int h = h0 - 1 + row;
          int w = col - 1;
          v8bf v;
#pragma unroll
          for (int z = 0; z < 8; ++z) v[z] = (__bf16)0.f;
          if ((unsigned)h < (unsigned)HH && (unsigned)w < (unsigned)WW)
            v = *(const v8bf*)(yb + ((size_t)h * WW + w) * CH + co8 * 8);
          *(v8bf*)&lds[(row * 58 + col) * LDS_CSTR + co8 * 8] = v;
        }
      }
    }
    __syncthreads();

    // ---- implicit GEMM over 9 taps x 2 K-chunks, fully unrolled ----
    v8f acc[4];
#pragma unroll
    for (int jj = 0; jj < 4; ++jj)
#pragma unroll
      for (int vg = 0; vg < 8; ++vg) acc[jj][vg] = 0.f;

    const __bf16* wf =
        wpack + ((((size_t)e * 2 + (LAYER - 1)) * 4 + m) * 9) * 2 * 512 + lane * 16;
#pragma unroll
    for (int t9 = 0; t9 < 9; ++t9) {
      const int r = t9 / 3, s = t9 % 3;
#pragma unroll
      for (int q = 0; q < 2; ++q) {
        v16bf a = *(const v16bf*)(wf + (t9 * 2 + q) * 512);
        const int tapOff = (r * 58 + s) * LDS_CSTR + q * 32;   // constant
#pragma unroll
        for (int jj = 0; jj < 4; ++jj) {
          v16bf bm = *(const v16bf*)(&lds[ldsBase[jj] + tapOff]);
          acc[jj] = __builtin_amdgcn_wmma_f32_16x16x32_bf16(
              false, a, false, bm, (short)0, acc[jj], false, false);
        }
      }
    }

    // ---- epilogue ----
    if (LAYER == 1) {
      __bf16* yb = y1 + (size_t)(b * TOPK + k) * HWSZ * CH;
#pragma unroll
      for (int jj = 0; jj < 4; ++jj) {
        if (pVal[jj]) {
          v8bf pk;
#pragma unroll
          for (int vg = 0; vg < 8; ++vg) {
            float v = acc[jj][vg] * bn_s[e * CH + cob + vg] + bn_b[e * CH + cob + vg];
            v = v > 0.f ? v : 0.f;
            pk[vg] = (__bf16)v;
          }
          *(v8bf*)(yb + (((size_t)(h0 + pRow[jj]) * WW + pCol[jj]) * CH + cob)) = pk;
        }
      }
    } else {
      const float gwk = sel_wt[b * 2 + k];
#pragma unroll
      for (int jj = 0; jj < 4; ++jj) {
#pragma unroll
        for (int vg = 0; vg < 8; ++vg) {
          float v = acc[jj][vg] * bn_s[e * CH + cob + vg] + bn_b[e * CH + cob + vg];
          v += xres[jj][vg];
          v = v > 0.f ? v : 0.f;
          res[jj][vg] += gwk * v;
        }
      }
    }
  }

  if (LAYER == 2) {
#pragma unroll
    for (int jj = 0; jj < 4; ++jj) {
      if (pVal[jj]) {
        int hw = (h0 + pRow[jj]) * WW + pCol[jj];
#pragma unroll
        for (int vg = 0; vg < 8; ++vg)
          out[((size_t)b * CH + cob + vg) * HWSZ + hw] = res[jj][vg];
      }
    }
  }
}

// ---------------------------------------------------------------------------
extern "C" void kernel_launch(void* const* d_in, const int* in_sizes, int n_in,
                              void* d_out, int out_size, void* d_ws, size_t ws_size,
                              hipStream_t stream)
{
  const float* x       = (const float*)d_in[0];
  const float* gate_w  = (const float*)d_in[1];
  const float* gate_b  = (const float*)d_in[2];
  const float* conv1_w = (const float*)d_in[3];
  const float* bn1_s   = (const float*)d_in[4];
  const float* bn1_b   = (const float*)d_in[5];
  const float* conv2_w = (const float*)d_in[6];
  const float* bn2_s   = (const float*)d_in[7];
  const float* bn2_b   = (const float*)d_in[8];
  float* out = (float*)d_out;

  char* ws = (char*)d_ws;
  __bf16* wpack = (__bf16*)ws;
  __bf16* y1    = (__bf16*)(ws + WPACK_BYTES);
  int*    sel_i = (int*)  (ws + WPACK_BYTES + Y1_BYTES);
  float*  sel_w = (float*)(ws + WPACK_BYTES + Y1_BYTES + 512);

  gate_kernel<<<BATCH, 256, 0, stream>>>(x, gate_w, gate_b,
                                         out + OUT_ELEMS, sel_i, sel_w);
  repack_kernel<<<(WPACK_ELEMS + 255) / 256, 256, 0, stream>>>(conv1_w, conv2_w, wpack);
  moe_conv_kernel<1><<<BATCH * TOPK * TILES, 256, 0, stream>>>(
      x, bn1_s, bn1_b, wpack, y1, sel_i, sel_w, nullptr);
  moe_conv_kernel<2><<<BATCH * TILES, 256, 0, stream>>>(
      x, bn2_s, bn2_b, wpack, y1, sel_i, sel_w, out);
}
